// GAT_4698694222360
// MI455X (gfx1250) — matrix operand order repaired and checked by reading, not compiled, not run on hardware
//
#include <hip/hip_runtime.h>
#include <hip/hip_bf16.h>

// ---------------------------------------------------------------------------
// Types for WMMA (CDNA5 / gfx1250, wave32)
// ---------------------------------------------------------------------------
typedef __attribute__((ext_vector_type(16))) __bf16      v16bf;
typedef __attribute__((ext_vector_type(8)))  float       v8f;
typedef __attribute__((ext_vector_type(4)))  unsigned int u32x4;

#define NEG_SLOPE 0.2f
#define CDIV(a, b) (((a) + (b) - 1) / (b))

// ---------------------------------------------------------------------------
// Helpers
// ---------------------------------------------------------------------------
__device__ __forceinline__ unsigned short f2bf(float x) {
  unsigned u = __float_as_uint(x);
  u += 0x7FFFu + ((u >> 16) & 1u);          // round-to-nearest-even
  return (unsigned short)(u >> 16);
}

// Order-preserving float -> uint key (for atomicMax on floats incl. negatives)
__device__ __forceinline__ unsigned fkey(float x) {
  unsigned u = __float_as_uint(x);
  return (u & 0x80000000u) ? ~u : (u | 0x80000000u);
}
__device__ __forceinline__ float unfkey(unsigned k) {
  unsigned u = (k & 0x80000000u) ? (k ^ 0x80000000u) : ~k;
  return __uint_as_float(u);
}

__device__ __forceinline__ float leaky(float v) {
  return v >= 0.f ? v : NEG_SLOPE * v;
}

// ---------------------------------------------------------------------------
// Utility kernels
// ---------------------------------------------------------------------------
__global__ void k_fill_u32(unsigned* __restrict__ p, unsigned v, int n) {
  int i = blockIdx.x * blockDim.x + threadIdx.x;
  if (i < n) p[i] = v;
}

__global__ void k_f32_to_bf16_pad(const float* __restrict__ in,
                                  unsigned short* __restrict__ out,
                                  int rows, int rowspad, int cols) {
  int i = blockIdx.x * blockDim.x + threadIdx.x;
  if (i >= rowspad * cols) return;
  int r = i / cols;
  float x = (r < rows) ? in[i] : 0.f;
  out[i] = f2bf(x);
}

__global__ void k_f32_to_bf16(const float* __restrict__ in,
                              unsigned short* __restrict__ out, int n) {
  int i = blockIdx.x * blockDim.x + threadIdx.x;
  if (i < n) out[i] = f2bf(in[i]);
}

__global__ void k_elu_to_bf16_pad(const float* __restrict__ in,
                                  unsigned short* __restrict__ out,
                                  int rows, int rowspad, int cols) {
  int i = blockIdx.x * blockDim.x + threadIdx.x;
  if (i >= rowspad * cols) return;
  int r = i / cols;
  float y = 0.f;
  if (r < rows) {
    float x = in[i];
    y = (x > 0.f) ? x : (__expf(x) - 1.f);   // jax.nn.elu
  }
  out[i] = f2bf(y);
}

// ---------------------------------------------------------------------------
// bf16 GEMM on the WMMA pipe: C[Mpad,Ncols] = A[Mpad,K](bf16) @ B[K,Ncols](bf16)
// One wave per 16x16 output tile; K consumed 32 at a time via
// v_wmma_f32_16x16x32_bf16 (f32 accumulate).
//
// A-tile layout (ISA 7.12.2, 16-bit A 16x32): lane L holds row M=L&15;
//   lanes 0-15: elems 0-7 = K kg..kg+7, elems 8-15 = K kg+16..kg+23, kg=0
//   lanes 16-31: same with kg=8.  Both halves are 16B-contiguous in memory.
// B-tile layout: lane&15 = N column; same K order as A (kg per lane half),
//   16 K values per lane, each a separate load at stride Ncols.
// D layout: lane&15 = N; VGPR v -> M = v + 8*(lane>>4).
// ---------------------------------------------------------------------------
__global__ void k_gemm_bf16_wmma(const unsigned short* __restrict__ A,
                                 const unsigned short* __restrict__ B,
                                 float* __restrict__ C,
                                 int K, int Ncols, int total_tiles) {
  int wave = blockIdx.x * (blockDim.x >> 5) + (threadIdx.x >> 5);
  if (wave >= total_tiles) return;           // whole-wave uniform exit
  int lane = threadIdx.x & 31;

  int tilesN = Ncols >> 4;
  int tm = wave / tilesN;
  int tn = wave - tm * tilesN;

  int arow = tm * 16 + (lane & 15);
  int kg   = (lane >> 4) * 8;
  const unsigned short* Arow = A + (size_t)arow * K;

  v8f acc = {};
  for (int k0 = 0; k0 < K; k0 += 32) {
    union { u32x4 u[2]; v16bf v; } a, b;
    // A: two 16-byte contiguous chunks per lane
    a.u[0] = *(const u32x4*)(Arow + k0 + kg);
    a.u[1] = *(const u32x4*)(Arow + k0 + kg + 16);
    // B: lane&15 = N column; elems 0-7 = K kg..kg+7, elems 8-15 = K kg+16..kg+23
    const unsigned short* Bcol = B + (size_t)(k0 + kg) * Ncols + tn * 16 + (lane & 15);
    for (int s = 0; s < 16; ++s)
      ((unsigned short*)&b)[s] = Bcol[(size_t)((s & 7) + (s >> 3) * 16) * Ncols];
    acc = __builtin_amdgcn_wmma_f32_16x16x32_bf16(
        /*neg_a=*/false, a.v, /*neg_b=*/false, b.v,
        /*c_mod=*/(short)0, acc, /*reuse_a=*/false, /*reuse_b=*/false);
  }

  int rbase = tm * 16 + 8 * (lane >> 4);
  int c0    = tn * 16 + (lane & 15);
#pragma unroll
  for (int v = 0; v < 8; ++v)
    C[(size_t)(rbase + v) * Ncols + c0] = acc[v];
}

// ---------------------------------------------------------------------------
// Per-node attention projections: es[n,h] = <z[n,h,:], a_src[h,:]>, same for ed
// ---------------------------------------------------------------------------
__global__ void k_dots(const float* __restrict__ z,
                       const float* __restrict__ a_src,
                       const float* __restrict__ a_dst,
                       float* __restrict__ es, float* __restrict__ ed,
                       int N, int H, int D) {
  int i = blockIdx.x * blockDim.x + threadIdx.x;
  if (i >= N * H) return;
  int n = i / H, h = i - n * H;
  const float* zp = z + (size_t)n * (H * D) + h * D;
  const float* as = a_src + h * D;
  const float* ad = a_dst + h * D;
  float s = 0.f, d = 0.f;
  for (int k = 0; k < D; ++k) { s += zp[k] * as[k]; d += zp[k] * ad[k]; }
  es[i] = s;
  ed[i] = d;
}

// ---------------------------------------------------------------------------
// Edge pass 1: segment max of leaky-relu logits (via order-preserving keys)
// ---------------------------------------------------------------------------
__global__ void k_edge_max(const int* __restrict__ src, const int* __restrict__ dst,
                           const float* __restrict__ es, const float* __restrict__ ed,
                           unsigned* __restrict__ mkey, int E, int H) {
  int i = blockIdx.x * blockDim.x + threadIdx.x;
  if (i >= E * H) return;
  int e = i / H, h = i - e * H;
  int s = src[e], d = dst[e];
  float v = leaky(es[s * H + h] + ed[d * H + h]);
  atomicMax(&mkey[d * H + h], fkey(v));
}

// ---------------------------------------------------------------------------
// Edge pass 2: segment sum of exp(e - m)
// ---------------------------------------------------------------------------
__global__ void k_edge_sum(const int* __restrict__ src, const int* __restrict__ dst,
                           const float* __restrict__ es, const float* __restrict__ ed,
                           const unsigned* __restrict__ mkey,
                           float* __restrict__ ssum, int E, int H) {
  int i = blockIdx.x * blockDim.x + threadIdx.x;
  if (i >= E * H) return;
  int e = i / H, h = i - e * H;
  int s = src[e], d = dst[e];
  float v = leaky(es[s * H + h] + ed[d * H + h]);
  float m = unfkey(mkey[d * H + h]);
  atomicAdd(&ssum[d * H + h], __expf(v - m));
}

// ---------------------------------------------------------------------------
// Edge pass 3: out[dst] += alpha * z[src].  One block per edge, F=H*D threads.
// ---------------------------------------------------------------------------
__global__ void k_edge_agg(const int* __restrict__ src, const int* __restrict__ dst,
                           const float* __restrict__ es, const float* __restrict__ ed,
                           const unsigned* __restrict__ mkey,
                           const float* __restrict__ ssum,
                           const float* __restrict__ z,
                           float* __restrict__ out, int H, int D) {
  int e = blockIdx.x;
  int t = threadIdx.x;              // 0 .. H*D-1
  int h = t / D;
  int F = H * D;
  int s = src[e], d = dst[e];
  float v = leaky(es[s * H + h] + ed[d * H + h]);
  float m = unfkey(mkey[d * H + h]);
  float alpha = __expf(v - m) / ssum[d * H + h];
  atomicAdd(&out[(size_t)d * F + t], alpha * z[(size_t)s * F + t]);
}

// ---------------------------------------------------------------------------
// Host-side orchestration
// ---------------------------------------------------------------------------
extern "C" void kernel_launch(void* const* d_in, const int* in_sizes, int n_in,
                              void* d_out, int out_size, void* d_ws, size_t ws_size,
                              hipStream_t stream) {
  const float* h   = (const float*)d_in[0];
  const float* W1  = (const float*)d_in[1];
  const float* a1s = (const float*)d_in[2];
  const float* a1d = (const float*)d_in[3];
  const float* W2  = (const float*)d_in[4];
  const float* a2s = (const float*)d_in[5];
  const float* a2d = (const float*)d_in[6];
  const int*   src = (const int*)d_in[7];
  const int*   dst = (const int*)d_in[8];

  const int IN_DIM = 128, H1 = 8, D1 = 16, F1 = H1 * D1 /*128*/, OUT = 32;
  const int N = in_sizes[0] / IN_DIM;
  const int E = in_sizes[7];
  const int Npad = (N + 15) & ~15;

  // ---- carve workspace -----------------------------------------------------
  char* base = (char*)d_ws;
  size_t off = 0;
  auto carve = [&](size_t bytes) -> void* {
    void* p = base + off;
    off += (bytes + 255) & ~(size_t)255;
    return p;
  };
  unsigned short* hb  = (unsigned short*)carve((size_t)Npad * IN_DIM * 2);
  unsigned short* w1b = (unsigned short*)carve((size_t)IN_DIM * F1 * 2);
  unsigned short* w2b = (unsigned short*)carve((size_t)F1 * OUT * 2);
  float*    z1   = (float*)carve((size_t)Npad * F1 * 4);
  float*    es1  = (float*)carve((size_t)N * H1 * 4);
  float*    ed1  = (float*)carve((size_t)N * H1 * 4);
  unsigned* mk1  = (unsigned*)carve((size_t)N * H1 * 4);
  float*    ss1  = (float*)carve((size_t)N * H1 * 4);
  float*    agg1 = (float*)carve((size_t)N * F1 * 4);
  unsigned short* h1b = (unsigned short*)carve((size_t)Npad * F1 * 2);
  float*    z2   = (float*)carve((size_t)Npad * OUT * 4);
  float*    es2  = (float*)carve((size_t)N * 4);
  float*    ed2  = (float*)carve((size_t)N * 4);
  unsigned* mk2  = (unsigned*)carve((size_t)N * 4);
  float*    ss2  = (float*)carve((size_t)N * 4);
  float*    out  = (float*)d_out;  // [N, OUT] f32

  const int B = 256;

  // ---- zero accumulators (deterministic across graph replays) -------------
  k_fill_u32<<<CDIV(N * F1, B), B, 0, stream>>>((unsigned*)agg1, 0u, N * F1);
  k_fill_u32<<<CDIV(N * H1, B), B, 0, stream>>>(mk1, 0u, N * H1);
  k_fill_u32<<<CDIV(N * H1, B), B, 0, stream>>>((unsigned*)ss1, 0u, N * H1);
  k_fill_u32<<<CDIV(N, B), B, 0, stream>>>(mk2, 0u, N);
  k_fill_u32<<<CDIV(N, B), B, 0, stream>>>((unsigned*)ss2, 0u, N);
  k_fill_u32<<<CDIV(N * OUT, B), B, 0, stream>>>((unsigned*)out, 0u, N * OUT);

  // ---- bf16 conversions ----------------------------------------------------
  k_f32_to_bf16_pad<<<CDIV(Npad * IN_DIM, B), B, 0, stream>>>(h, hb, N, Npad, IN_DIM);
  k_f32_to_bf16<<<CDIV(IN_DIM * F1, B), B, 0, stream>>>(W1, w1b, IN_DIM * F1);
  k_f32_to_bf16<<<CDIV(F1 * OUT, B), B, 0, stream>>>(W2, w2b, F1 * OUT);

  // ---- layer 1: GEMM (WMMA) + attention -----------------------------------
  {
    int tiles = (Npad / 16) * (F1 / 16);
    k_gemm_bf16_wmma<<<CDIV(tiles, 8), 256, 0, stream>>>(hb, w1b, z1, IN_DIM, F1, tiles);
  }
  k_dots<<<CDIV(N * H1, B), B, 0, stream>>>(z1, a1s, a1d, es1, ed1, N, H1, D1);
  k_edge_max<<<CDIV(E * H1, B), B, 0, stream>>>(src, dst, es1, ed1, mk1, E, H1);
  k_edge_sum<<<CDIV(E * H1, B), B, 0, stream>>>(src, dst, es1, ed1, mk1, ss1, E, H1);
  k_edge_agg<<<E, F1, 0, stream>>>(src, dst, es1, ed1, mk1, ss1, z1, agg1, H1, D1);

  // ---- ELU + re-quantize to bf16 ------------------------------------------
  k_elu_to_bf16_pad<<<CDIV(Npad * F1, B), B, 0, stream>>>(agg1, h1b, N, Npad, F1);

  // ---- layer 2: GEMM (WMMA) + attention -----------------------------------
  {
    int tiles = (Npad / 16) * (OUT / 16);
    k_gemm_bf16_wmma<<<CDIV(tiles, 8), 256, 0, stream>>>(h1b, w2b, z2, F1, OUT, tiles);
  }
  k_dots<<<CDIV(N, B), B, 0, stream>>>(z2, a2s, a2d, es2, ed2, N, 1, OUT);
  k_edge_max<<<CDIV(E, B), B, 0, stream>>>(src, dst, es2, ed2, mk2, E, 1);
  k_edge_sum<<<CDIV(E, B), B, 0, stream>>>(src, dst, es2, ed2, mk2, ss2, E, 1);
  k_edge_agg<<<E, OUT, 0, stream>>>(src, dst, es2, ed2, mk2, ss2, z2, out, 1, OUT);

  (void)n_in; (void)out_size; (void)ws_size;
}
